// MPNN_25546465476713
// MI455X (gfx1250) — compile-verified, matrix operand order checked by reading, not run on hardware
//
#include <hip/hip_runtime.h>
#include <hip/hip_bf16.h>

// ---------------------------------------------------------------------------
// GCN forward on MI455X (gfx1250, wave32):
//   deg -> dinv -> [GEMM(WMMA f16) + dinv-scale] -> edge scatter (f32 atomics)
//   -> relu(dinv*acc + b) -> repeat -> fused pool+classifier.
// Weights pre-packed into WMMA fragment-major f16 so B loads are b128.
// ---------------------------------------------------------------------------

typedef __attribute__((ext_vector_type(16))) _Float16 v16h;
typedef __attribute__((ext_vector_type(8)))  float    v8f;

#define D 128   // feature dim == hidden dim
#define NUM_GRAPHS 64

// Fragment K-mapping per CDNA5 ISA (16-bit A/B, 16x32):
// lane L -> m/n = L&15, half = L>>4; packed element e (0..15):
//   K = 16*(e>=8) + 8*half + (e&7)   [two contiguous runs of 8]
__device__ __forceinline__ int frag_k(int e, int half) {
    int j = e >> 1;
    return ((j >> 2) << 4) + (half << 3) + ((j & 3) << 1) + (e & 1);
}

// ---------------- weight packing: W[K,128] f32 -> fragment-major f16 -------
// P[((kt*8+nt)*32 + lane)*16 + e] = W[(kt*32 + frag_k(e, lane>>4))*128 + nt*16 + (lane&15)]

__global__ void pack_weights(const float* __restrict__ W1, const float* __restrict__ W2,
                             _Float16* __restrict__ P1, _Float16* __restrict__ P2) {
    int idx = blockIdx.x * blockDim.x + threadIdx.x;
    if (idx >= D * D) return;
    const int e    = idx & 15;
    const int lane = (idx >> 4) & 31;
    const int nt   = (idx >> 9) & 7;
    const int kt   = idx >> 12;
    const int K    = kt * 32 + frag_k(e, lane >> 4);
    const int col  = nt * 16 + (lane & 15);
    P1[idx] = (_Float16)W1[K * D + col];
    P2[idx] = (_Float16)W2[K * D + col];
}

// ---------------- small utility kernels ----------------

__global__ void init_deg(float* __restrict__ deg, int N) {
    int i = blockIdx.x * blockDim.x + threadIdx.x;
    if (i < N) deg[i] = 1.0f;   // self-loop contributes 1 to every node's degree
}

__global__ void deg_scatter(const int* __restrict__ cols, float* __restrict__ deg, int E) {
    int i = blockIdx.x * blockDim.x + threadIdx.x;
    if (i < E) unsafeAtomicAdd(&deg[cols[i]], 1.0f);
}

__global__ void finish_dinv(float* __restrict__ deg, int N) {
    int i = blockIdx.x * blockDim.x + threadIdx.x;
    if (i < N) deg[i] = rsqrtf(deg[i]);   // deg >= 1 always (self-loops)
}

// ---------------- WMMA GEMM: T/Acc = (X @ W) * dinv[:,None] ----------------
// One wave owns a 16-row strip; 8 waves per block cover 128 rows.
// A fragment: 4 aligned float4 loads (two contiguous K-runs of 8) + cvt.
// B fragment: one aligned v16h load (2x global_load_b128) from packed weights.

__global__ __launch_bounds__(256)
void gemm_scaled(const float* __restrict__ X, const _Float16* __restrict__ Wpk,
                 const float* __restrict__ dinv,
                 float* __restrict__ Tout, float* __restrict__ Acc, int N) {
    const int wave = threadIdx.x >> 5;
    const int lane = threadIdx.x & 31;
    const int rowBase = blockIdx.x * 128 + wave * 16;
    if (rowBase >= N) return;                        // wave-uniform exit (WMMA-safe)

    const int m    = lane & 15;
    const int half = lane >> 4;
    const int arow = (rowBase + m < N) ? (rowBase + m) : (N - 1);  // clamped read
    const v16h* __restrict__ Bp = reinterpret_cast<const v16h*>(Wpk);

    v8f acc[8] = {};

    #pragma unroll
    for (int kt = 0; kt < 4; ++kt) {
        const int k0 = kt * 32;
        const float* rowp = X + (size_t)arow * D + k0 + (half << 3);
        const float4 fa0 = *reinterpret_cast<const float4*>(rowp);       // K rel 0..3
        const float4 fa1 = *reinterpret_cast<const float4*>(rowp + 4);   // K rel 4..7
        const float4 fb0 = *reinterpret_cast<const float4*>(rowp + 16);  // K rel 16..19
        const float4 fb1 = *reinterpret_cast<const float4*>(rowp + 20);  // K rel 20..23

        v16h a;
        a[0]  = (_Float16)fa0.x; a[1]  = (_Float16)fa0.y;
        a[2]  = (_Float16)fa0.z; a[3]  = (_Float16)fa0.w;
        a[4]  = (_Float16)fa1.x; a[5]  = (_Float16)fa1.y;
        a[6]  = (_Float16)fa1.z; a[7]  = (_Float16)fa1.w;
        a[8]  = (_Float16)fb0.x; a[9]  = (_Float16)fb0.y;
        a[10] = (_Float16)fb0.z; a[11] = (_Float16)fb0.w;
        a[12] = (_Float16)fb1.x; a[13] = (_Float16)fb1.y;
        a[14] = (_Float16)fb1.z; a[15] = (_Float16)fb1.w;

        #pragma unroll
        for (int nt = 0; nt < 8; ++nt) {
            const v16h b = Bp[((kt * 8 + nt) << 5) + lane];
            acc[nt] = __builtin_amdgcn_wmma_f32_16x16x32_f16(
                false, a, false, b, (short)0, acc[nt], false, false);
        }
    }

    // Epilogue: element v of acc[nt] is (row = rowBase + v + 8*half, col = nt*16 + m).
    #pragma unroll
    for (int v = 0; v < 8; ++v) {
        const int row = rowBase + v + (half << 3);
        if (row < N) {
            const float s = dinv[row];
            #pragma unroll
            for (int nt = 0; nt < 8; ++nt) {
                const float val = acc[nt][v] * s;
                Tout[(size_t)row * D + nt * 16 + m] = val;  // message source
                Acc [(size_t)row * D + nt * 16 + m] = val;  // self-loop init
            }
        }
    }
}

// ---------------- edge scatter: Acc[col] += T[row] (float4 gather, f32 atomics) ---

__global__ void scatter_edges(const float* __restrict__ T, float* __restrict__ Acc,
                              const int* __restrict__ rows, const int* __restrict__ cols,
                              int E) {
    const int total = E * 32;                         // 32 float4-chunks per edge
    for (int i = blockIdx.x * blockDim.x + threadIdx.x; i < total;
         i += gridDim.x * blockDim.x) {
        const int e  = i >> 5;
        const int c4 = (i & 31) << 2;
        const int r = rows[e], c = cols[e];
        const float4 v = *reinterpret_cast<const float4*>(T + (size_t)r * D + c4);
        float* dst = Acc + (size_t)c * D + c4;
        unsafeAtomicAdd(dst + 0, v.x);
        unsafeAtomicAdd(dst + 1, v.y);
        unsafeAtomicAdd(dst + 2, v.z);
        unsafeAtomicAdd(dst + 3, v.w);
    }
}

// ---------------- activation: H = relu(dinv[i] * Acc + b[c]) ----------------

__global__ void act_kernel(const float* __restrict__ Acc, const float* __restrict__ dinv,
                           const float* __restrict__ b, float* __restrict__ Hout,
                           int N) {
    const int total = N * D;
    for (int i = blockIdx.x * blockDim.x + threadIdx.x; i < total;
         i += gridDim.x * blockDim.x) {
        const float v = dinv[i >> 7] * Acc[i] + b[i & (D - 1)];
        Hout[i] = v > 0.0f ? v : 0.0f;
    }
}

// ---------------- fused mean-pool + classifier ----------------

__global__ void init_pool(float* __restrict__ gsum, float* __restrict__ gcnt) {
    int i = threadIdx.x;
    if (i < NUM_GRAPHS) { gsum[i] = 0.0f; gcnt[i] = 0.0f; }
}

__global__ void pool_dot(const float* __restrict__ H2, const float* __restrict__ Wc,
                         const int* __restrict__ batch,
                         float* __restrict__ gsum, float* __restrict__ gcnt, int N) {
    const int wid  = (blockIdx.x * blockDim.x + threadIdx.x) >> 5;  // node id
    const int lane = threadIdx.x & 31;
    if (wid >= N) return;
    const float4 h = *reinterpret_cast<const float4*>(H2 + (size_t)wid * D + lane * 4);
    const float4 w = *reinterpret_cast<const float4*>(Wc + lane * 4);
    float d = h.x * w.x + h.y * w.y + h.z * w.z + h.w * w.w;
    #pragma unroll
    for (int off = 16; off > 0; off >>= 1) d += __shfl_down(d, off, 32);
    if (lane == 0) {
        const int g = batch[wid];
        unsafeAtomicAdd(&gsum[g], d);
        unsafeAtomicAdd(&gcnt[g], 1.0f);
    }
}

__global__ void final_kernel(const float* __restrict__ gsum, const float* __restrict__ gcnt,
                             const float* __restrict__ bc, float* __restrict__ out) {
    int g = threadIdx.x;
    if (g < NUM_GRAPHS) {
        float c = gcnt[g];
        out[g] = gsum[g] / (c > 1.0f ? c : 1.0f) + bc[0];
    }
}

// ---------------- launch ----------------

extern "C" void kernel_launch(void* const* d_in, const int* in_sizes, int n_in,
                              void* d_out, int out_size, void* d_ws, size_t ws_size,
                              hipStream_t stream) {
    const float* x     = (const float*)d_in[0];
    const int*   ei    = (const int*)  d_in[1];
    const int*   batch = (const int*)  d_in[2];
    const float* W1    = (const float*)d_in[3];
    const float* b1    = (const float*)d_in[4];
    const float* W2    = (const float*)d_in[5];
    const float* b2    = (const float*)d_in[6];
    const float* Wc    = (const float*)d_in[7];
    const float* bc    = (const float*)d_in[8];
    float* out = (float*)d_out;

    const int N = in_sizes[0] / D;       // 50000
    const int E = in_sizes[1] / 2;       // 800000
    const int* rows = ei;                // edge_index[0] = source
    const int* cols = ei + E;            // edge_index[1] = target

    // workspace carve-up (d_ws is 256B-aligned; all offsets stay 32B-aligned)
    const size_t NH = (size_t)N * D;
    float*     T    = (float*)d_ws;            // [N, D] messages / layer output
    float*     Acc  = T + NH;                  // [N, D] aggregation accumulator
    float*     dinv = Acc + NH;                // [N]    deg -> rsqrt(deg)
    _Float16*  W1p  = (_Float16*)(dinv + N);   // [D*D] packed f16 fragments
    _Float16*  W2p  = W1p + D * D;
    float*     gsum = (float*)(W2p + D * D);   // [64]
    float*     gcnt = gsum + NUM_GRAPHS;       // [64]

    const int nb_N   = (N + 255) / 256;
    const int nb_E   = (E + 255) / 256;
    const int nb_gem = (N + 127) / 128;
    const int nb_sc  = (E * 32 + 255) / 256;
    const int nb_act = (int)((NH + 255) / 256);
    const int nb_pl  = (N * 32 + 255) / 256;

    // weight packing + degree normalization
    pack_weights<<<(D * D + 255) / 256, 256, 0, stream>>>(W1, W2, W1p, W2p);
    init_deg<<<nb_N, 256, 0, stream>>>(dinv, N);
    deg_scatter<<<nb_E, 256, 0, stream>>>(cols, dinv, E);
    finish_dinv<<<nb_N, 256, 0, stream>>>(dinv, N);

    // layer 1
    gemm_scaled<<<nb_gem, 256, 0, stream>>>(x, W1p, dinv, T, Acc, N);
    scatter_edges<<<nb_sc, 256, 0, stream>>>(T, Acc, rows, cols, E);
    act_kernel<<<nb_act, 256, 0, stream>>>(Acc, dinv, b1, T, N);   // T <- h1

    // layer 2 (GEMM in place over T: each wave reads only its own rows)
    gemm_scaled<<<nb_gem, 256, 0, stream>>>(T, W2p, dinv, T, Acc, N);
    scatter_edges<<<nb_sc, 256, 0, stream>>>(T, Acc, rows, cols, E);
    act_kernel<<<nb_act, 256, 0, stream>>>(Acc, dinv, b2, T, N);   // T <- h2

    // pooled classifier
    init_pool<<<1, 64, 0, stream>>>(gsum, gcnt);
    pool_dot<<<nb_pl, 256, 0, stream>>>(T, Wc, batch, gsum, gcnt, N);
    final_kernel<<<1, 64, 0, stream>>>(gsum, gcnt, bc, out);
}
